// FusetViTBlock_69191923138813
// MI455X (gfx1250) — matrix operand order repaired and checked by reading, not compiled, not run on hardware
//
#include <hip/hip_runtime.h>
#include <hip/hip_bf16.h>

// ---------------------------------------------------------------------------
// MI455X (gfx1250) fused EfficientViT block.
// All big GEMMs (qkv/proj/inv/pw 1x1 convs, ~94% of the 60 GFLOP total) run
// through v_wmma_f32_16x16x32_bf16 (bf16 in, f32 accumulate). Everything fits
// in the 192MB L2, so the chain is compute bound and the bf16 WMMA path is the
// right ceiling. Weight tiles are DMA'd into LDS with the Tensor Data Mover
// (TENSOR_LOAD_TO_LDS + s_wait_tensorcnt), overlapping with the f32->bf16
// activation staging. dw-convs / grouped-pw / attention stay f32 VALU.
// ---------------------------------------------------------------------------

typedef __attribute__((ext_vector_type(16))) __bf16 v16bf;
typedef __attribute__((ext_vector_type(8)))  __bf16 v8bf;
typedef __attribute__((ext_vector_type(8)))  float  v8f;
typedef unsigned int v4u __attribute__((ext_vector_type(4)));
typedef int          v8i __attribute__((ext_vector_type(8)));
typedef int          v4i __attribute__((ext_vector_type(4)));

union Frag16 { v16bf v; v8bf h[2]; };

#if defined(__gfx1250__) && __has_builtin(__builtin_amdgcn_tensor_load_to_lds)
#define HAS_TDM 1
#else
#define HAS_TDM 0
#endif

__device__ __forceinline__ float hswish_f(float x) {
    float c = fminf(fmaxf(x + 3.f, 0.f), 6.f);
    return x * c * (1.f / 6.f);
}

// round-to-nearest-even f32 -> bf16
__device__ __forceinline__ __bf16 f2bf(float f) {
    union { float f; unsigned u; } in; in.f = f;
    unsigned r = (in.u + 0x7FFFu + ((in.u >> 16) & 1u)) >> 16;
    union { unsigned short s; __bf16 b; } out; out.s = (unsigned short)r;
    return out.b;
}

#if HAS_TDM
// TDM: load a 128-row x 32-col bf16 tile (row stride Ktot elems) into LDS,
// padding each 64B row by 16B so the LDS row stride is 80B (= LDA*2).
__device__ __forceinline__ void tdm_load_tile_a(const __bf16* gsrc, unsigned lds_addr,
                                                int Ktot, int Mtot) {
    unsigned long long ga = (unsigned long long)(size_t)gsrc;
    v4u g0;
    g0.x = 1u;                                   // count=1, user descriptor
    g0.y = lds_addr;                             // lds_addr (bytes)
    g0.z = (unsigned)(ga & 0xffffffffu);         // global_addr[31:0]
    g0.w = (unsigned)(ga >> 32) | (2u << 30);    // global_addr[56:32] | type=2
    v8i g1;
    g1[0] = (1 << 16)      // data_size = 2B
          | (1 << 20)      // pad_enable
          | (3 << 22)      // pad_interval: every 16 DWORDs (64B)
          | (3 << 25);     // pad_amount: 4 DWORDs (16B)
    g1[1] = (Ktot & 0xffff) << 16;   // tensor_dim0[15:0]  (abar addr = 0)
    g1[2] = (Mtot & 0xffff) << 16;   // tensor_dim1[15:0]
    g1[3] = 32 << 16;                // tile_dim0 = 32
    g1[4] = 128;                     // tile_dim1 = 128, tile_dim2 = 0
    g1[5] = Ktot;                    // tensor_dim0_stride[31:0]
    g1[6] = 0;
    g1[7] = 0;
    v4i gz4 = {0, 0, 0, 0};
    v8i gz8 = {0, 0, 0, 0, 0, 0, 0, 0};
    // clang-23 / therock arity: (g0, g1, g2, g3, g4, cpol)
    __builtin_amdgcn_tensor_load_to_lds(g0, g1, gz4, gz4, gz8, 0);
}
#endif

__global__ __launch_bounds__(256) void cvt_bf16_kernel(const float* __restrict__ src,
                                                       __bf16* __restrict__ dst, int n) {
    int i = blockIdx.x * 256 + threadIdx.x;
    if (i < n) dst[i] = f2bf(src[i]);
}

// ---------------------------------------------------------------------------
// Generic bf16 WMMA GEMM over activations laid out [B][K][1024] (1024 pixels).
//   Out[b][m][n] = sum_k W[m][k] * Bact[b][k][n]   (+ fused epilogue)
// Block: 256 threads (8 waves). Tile: M=128, N=128, K step 32.
// EPI: 0 none | 1 BN+residual | 2 bias+hswish | 3 BN+residual then Out (+)=.
// FUSE_DW: B-tile = hswish(dwconv3x3(Bact) + bias) computed during staging.
// ---------------------------------------------------------------------------
enum { EPI_NONE = 0, EPI_BN_RES = 1, EPI_BIAS_HSWISH = 2, EPI_BN_RES_OUT = 3 };

template <int EPI, bool FUSE_DW, bool ACCUM>
__global__ __launch_bounds__(256) void gemm_bf16_kernel(
    const __bf16* __restrict__ W,    // [Mtot x Ktot] bf16 row-major
    const float*  __restrict__ Bact, // [16][Ktot][1024] f32
    float*        __restrict__ Out,  // [16][Mtot][1024] f32
    int Mtot, int Ktot,
    const float* __restrict__ p0, const float* __restrict__ p1,
    const float* __restrict__ p2, const float* __restrict__ p3,
    const float* __restrict__ resid,
    const float* __restrict__ dww, const float* __restrict__ dwb) {
    constexpr int LDA = 40;  // 32 k + pad (80B rows: 16B-aligned chunks, bank skew)
    constexpr int LDB = 40;
    __shared__ __align__(16) __bf16 ldsA[128 * LDA];
    __shared__ __align__(16) __bf16 ldsB[128 * LDB];
    __shared__ float cw[32 * 9];
    __shared__ float cb[32];

    const int t    = threadIdx.x;
    const int lane = t & 31;
    const int wave = t >> 5;
    const bool hi  = lane >= 16;
    const int lm   = lane & 15;

    const int mBase = blockIdx.x * 128;
    const int n0    = blockIdx.y * 128;
    const int b     = blockIdx.z;

    v8f cf[8];
#pragma unroll
    for (int j = 0; j < 8; ++j)
#pragma unroll
        for (int e = 0; e < 8; ++e) cf[j][e] = 0.f;

    const float* bsrc = Bact + (size_t)b * Ktot * 1024;

    for (int k0 = 0; k0 < Ktot; k0 += 32) {
        __syncthreads();
        // ---- stage A: 128 rows x 32 k ------------------------------------
#if HAS_TDM
        if (wave == 0) {
            tdm_load_tile_a(W + (size_t)mBase * Ktot + k0,
                            (unsigned)(size_t)&ldsA[0], Ktot, Mtot);
        }
#else
#pragma unroll 2
        for (int i = t; i < 128 * 4; i += 256) {
            int r = i >> 2, seg = i & 3;
            const uint4* src = (const uint4*)(W + (size_t)(mBase + r) * Ktot + k0 + seg * 8);
            *(uint4*)&ldsA[r * LDA + seg * 8] = *src;
        }
#endif
        if (k0 + 32 < Ktot)  // L2 prefetch of next weight slab
            __builtin_prefetch(W + (size_t)(mBase + (t >> 1)) * Ktot + k0 + 32 + (t & 1) * 16, 0, 0);

        // ---- stage B transposed [n][k] as bf16 ----------------------------
        if (!FUSE_DW) {
            for (int i = t; i < 32 * 128; i += 256) {
                int n = i & 127, k = i >> 7;
                float v = bsrc[(size_t)(k0 + k) * 1024 + n0 + n];
                ldsB[n * LDB + k] = f2bf(v);
            }
        } else {
            if (t < 32 * 9) cw[t] = dww[(size_t)(k0 + t / 9) * 9 + t % 9];
            if (t < 32)     cb[t] = dwb[k0 + t];
            __syncthreads();
            for (int i = t; i < 32 * 128; i += 256) {
                int n = i & 127, k = i >> 7;
                int p = n0 + n, y = p >> 5, x = p & 31;
                const float* src = bsrc + (size_t)(k0 + k) * 1024;
                const float* wk  = &cw[k * 9];
                float s = cb[k];
#pragma unroll
                for (int dy = 0; dy < 3; ++dy) {
                    int yy = y + dy - 1;
                    if ((unsigned)yy < 32u) {
#pragma unroll
                        for (int dx = 0; dx < 3; ++dx) {
                            int xx = x + dx - 1;
                            if ((unsigned)xx < 32u) s += src[yy * 32 + xx] * wk[dy * 3 + dx];
                        }
                    }
                }
                ldsB[n * LDB + k] = f2bf(hswish_f(s));
            }
        }
#if HAS_TDM
        __builtin_amdgcn_s_wait_tensorcnt(0);  // no-op for waves that didn't issue
#endif
        __syncthreads();

        // ---- WMMA: one A frag per wave, 8 B frags -------------------------
        // A layout (ISA 16-bit 16x32): lanes0-15 K=0..7 & 16..23, lanes16-31 K=8..15 & 24..31
        Frag16 a;
        const int arow = wave * 16 + lm;
        const int ka   = hi ? 8 : 0;
        a.h[0] = *(const v8bf*)&ldsA[arow * LDA + ka];
        a.h[1] = *(const v8bf*)&ldsA[arow * LDA + ka + 16];
#pragma unroll
        for (int j = 0; j < 8; ++j) {
            Frag16 bb;
            const int brow = j * 16 + lm;       // column n of B
            const int kb   = hi ? 8 : 0;        // lanes0-15 K=0..7 & 16..23, lanes16-31 K=8..15 & 24..31
            bb.h[0] = *(const v8bf*)&ldsB[brow * LDB + kb];
            bb.h[1] = *(const v8bf*)&ldsB[brow * LDB + kb + 16];
            cf[j] = __builtin_amdgcn_wmma_f32_16x16x32_bf16(
                false, a.v, false, bb.v, (short)0, cf[j], false, false);
        }
    }

    // ---- epilogue ---------------------------------------------------------
#pragma unroll
    for (int j = 0; j < 8; ++j) {
#pragma unroll
        for (int v = 0; v < 8; ++v) {
            const int ch = mBase + wave * 16 + v + (hi ? 8 : 0);
            const int p  = n0 + j * 16 + lm;
            const size_t oidx = ((size_t)b * Mtot + ch) * 1024 + p;
            float val = cf[j][v];
            if (EPI == EPI_BN_RES || EPI == EPI_BN_RES_OUT) {
                float inv = p0[ch] / sqrtf(p3[ch] + 1e-5f);
                val = val * inv + (p1[ch] - p2[ch] * inv);
                val += resid[oidx];
            } else if (EPI == EPI_BIAS_HSWISH) {
                val = hswish_f(val + p0[ch]);
            }
            if (EPI == EPI_BN_RES_OUT) {
                if (ACCUM) Out[oidx] += val;
                else       Out[oidx]  = val;
            } else {
                Out[oidx] = val;
            }
        }
    }
}

// ---------------------------------------------------------------------------
// dwconv 5x5 (SAME, zero pad) + grouped 32x32 pointwise, per (group, 4-row strip, batch)
// ---------------------------------------------------------------------------
__global__ __launch_bounds__(256) void dw5_gpw_kernel(const float* __restrict__ qkvb,
                                                      const float* __restrict__ dw5,
                                                      const float* __restrict__ gpw,
                                                      float* __restrict__ pwb) {
    const int g = blockIdx.x, strip = blockIdx.y, b = blockIdx.z;
    const int y0 = strip * 4;
    __shared__ float dt[32][132];
    __shared__ float w5[32 * 25];
    __shared__ float wg[32][32];
    const int t = threadIdx.x;

    for (int i = t; i < 32 * 25; i += 256) w5[i] = dw5[(size_t)(g * 32 + i / 25) * 25 + i % 25];
    for (int i = t; i < 1024; i += 256)    wg[i >> 5][i & 31] = gpw[(size_t)g * 1024 + i];
    __syncthreads();

    const float* qc = qkvb + ((size_t)b * 768 + g * 32) * 1024;
    for (int i = t; i < 32 * 128; i += 256) {
        int ch = i >> 7, p = i & 127;
        int y = y0 + (p >> 5), x = p & 31;
        const float* src = qc + (size_t)ch * 1024;
        const float* wk  = &w5[ch * 25];
        float s = 0.f;
#pragma unroll
        for (int dy = 0; dy < 5; ++dy) {
            int yy = y + dy - 2;
            if ((unsigned)yy < 32u) {
#pragma unroll
                for (int dx = 0; dx < 5; ++dx) {
                    int xx = x + dx - 2;
                    if ((unsigned)xx < 32u) s += src[yy * 32 + xx] * wk[dy * 5 + dx];
                }
            }
        }
        dt[ch][p] = s;
    }
    __syncthreads();

    const int p  = t & 127;
    const int ob = (t >> 7) * 16;
    float acc[16];
#pragma unroll
    for (int o = 0; o < 16; ++o) acc[o] = 0.f;
    for (int i = 0; i < 32; ++i) {
        float dv = dt[i][p];
#pragma unroll
        for (int o = 0; o < 16; ++o) acc[o] += wg[ob + o][i] * dv;
    }
    const size_t base = ((size_t)b * 768 + g * 32 + ob) * 1024 + (size_t)y0 * 32 + p;
#pragma unroll
    for (int o = 0; o < 16; ++o) pwb[base + (size_t)o * 1024] = acc[o];
}

// ---------------------------------------------------------------------------
// ReLU linear attention per (head, batch): kv = [v;1] @ relu(k)^T  (33x32 over N=1024),
// out = kv @ relu(q), normalized by last row + 1e-15.
// ---------------------------------------------------------------------------
__global__ __launch_bounds__(256) void attn_kernel(const float* __restrict__ qkvb,
                                                   const float* __restrict__ pwb,
                                                   float* __restrict__ att) {
    const int h = blockIdx.x, b = blockIdx.y;
    const float* base = (h < 8) ? qkvb + ((size_t)b * 768 + h * 96) * 1024
                                : pwb  + ((size_t)b * 768 + (h - 8) * 96) * 1024;
    const float* qp = base;
    const float* kp = base + 32 * 1024;
    const float* vp = base + 64 * 1024;

    __shared__ float kb[32][129];
    __shared__ float vb[32][129];
    __shared__ float kv[33][32];
    const int t = threadIdx.x;

    float acc[5] = {0.f, 0.f, 0.f, 0.f, 0.f};
    for (int c0 = 0; c0 < 1024; c0 += 128) {
        __syncthreads();
        for (int i = t; i < 32 * 128; i += 256) {
            int ch = i >> 7, n = i & 127;
            kb[ch][n] = fmaxf(kp[(size_t)ch * 1024 + c0 + n], 0.f);
            vb[ch][n] = vp[(size_t)ch * 1024 + c0 + n];
        }
        __syncthreads();
#pragma unroll
        for (int pi = 0; pi < 5; ++pi) {
            int pr = t + 256 * pi;
            if (pr < 1056) {
                int d = pr >> 5, e = pr & 31;
                float s = acc[pi];
                if (d < 32) {
                    for (int n = 0; n < 128; ++n) s += vb[d][n] * kb[e][n];
                } else {
                    for (int n = 0; n < 128; ++n) s += kb[e][n];
                }
                acc[pi] = s;
            }
        }
    }
    __syncthreads();
#pragma unroll
    for (int pi = 0; pi < 5; ++pi) {
        int pr = t + 256 * pi;
        if (pr < 1056) kv[pr >> 5][pr & 31] = acc[pi];
    }
    __syncthreads();

    for (int n = t; n < 1024; n += 256) {
        float qv[32];
#pragma unroll
        for (int e = 0; e < 32; ++e) qv[e] = fmaxf(qp[(size_t)e * 1024 + n], 0.f);
        float den = 0.f;
#pragma unroll
        for (int e = 0; e < 32; ++e) den += kv[32][e] * qv[e];
        float rden = 1.f / (den + 1e-15f);
        for (int d = 0; d < 32; ++d) {
            float num = 0.f;
#pragma unroll
            for (int e = 0; e < 32; ++e) num += kv[d][e] * qv[e];
            att[((size_t)b * 512 + h * 32 + d) * 1024 + n] = num * rden;
        }
    }
}

// ---------------------------------------------------------------------------
extern "C" void kernel_launch(void* const* d_in, const int* in_sizes, int n_in,
                              void* d_out, int out_size, void* d_ws, size_t ws_size,
                              hipStream_t stream) {
    const float* x        = (const float*)d_in[0];
    const float* y        = (const float*)d_in[1];
    const float* qkv_w    = (const float*)d_in[2];
    const float* agg_dw_w = (const float*)d_in[3];
    const float* agg_pw_w = (const float*)d_in[4];
    const float* proj_w   = (const float*)d_in[5];
    const float* proj_g   = (const float*)d_in[6];
    const float* proj_b   = (const float*)d_in[7];
    const float* proj_m   = (const float*)d_in[8];
    const float* proj_v   = (const float*)d_in[9];
    const float* inv_w    = (const float*)d_in[10];
    const float* inv_b    = (const float*)d_in[11];
    const float* dwc_w    = (const float*)d_in[12];
    const float* dwc_b    = (const float*)d_in[13];
    const float* pw_w     = (const float*)d_in[14];
    const float* pw_g     = (const float*)d_in[15];
    const float* pw_b     = (const float*)d_in[16];
    const float* pw_m     = (const float*)d_in[17];
    const float* pw_v     = (const float*)d_in[18];

    char* ws = (char*)d_ws;
    size_t off = 0;
    auto take = [&](size_t bytes) -> char* {
        char* p = ws + off;
        off = (off + bytes + 255) & ~(size_t)255;
        return p;
    };

    __bf16* wq  = (__bf16*)take((size_t)768 * 256 * 2);
    __bf16* wpj = (__bf16*)take((size_t)256 * 512 * 2);
    __bf16* wiv = (__bf16*)take((size_t)1024 * 256 * 2);
    __bf16* wpw = (__bf16*)take((size_t)256 * 1024 * 2);
    float* qkvbuf = (float*)take((size_t)16 * 768 * 1024 * 4);
    float* pwbuf  = (float*)take((size_t)16 * 768 * 1024 * 4);
    float* attbuf = (float*)take((size_t)16 * 512 * 1024 * 4);
    float* t2buf  = (float*)take((size_t)16 * 256 * 1024 * 4);
    float* h1buf  = qkvbuf;  // alias: qkv/pw regions free once attention done

    auto cvt = [&](const float* s, __bf16* d, int n) {
        cvt_bf16_kernel<<<(n + 255) / 256, 256, 0, stream>>>(s, d, n);
    };
    cvt(qkv_w, wq, 768 * 256);
    cvt(proj_w, wpj, 256 * 512);
    cvt(inv_w, wiv, 1024 * 256);
    cvt(pw_w, wpw, 256 * 1024);

    float* out = (float*)d_out;

    for (int pass = 0; pass < 2; ++pass) {
        const float* t = pass ? y : x;

        // qkv = W_qkv @ t
        gemm_bf16_kernel<EPI_NONE, false, false><<<dim3(6, 8, 16), 256, 0, stream>>>(
            wq, t, qkvbuf, 768, 256,
            nullptr, nullptr, nullptr, nullptr, nullptr, nullptr, nullptr);

        // multi-scale branch: dw5x5 + grouped pw
        dw5_gpw_kernel<<<dim3(24, 8, 16), 256, 0, stream>>>(qkvbuf, agg_dw_w, agg_pw_w, pwbuf);

        // relu linear attention
        attn_kernel<<<dim3(16, 16), 256, 0, stream>>>(qkvbuf, pwbuf, attbuf);

        // t2 = t + BN(W_proj @ att)
        gemm_bf16_kernel<EPI_BN_RES, false, false><<<dim3(2, 8, 16), 256, 0, stream>>>(
            wpj, attbuf, t2buf, 256, 512,
            proj_g, proj_b, proj_m, proj_v, t, nullptr, nullptr);

        // h1 = hswish(W_inv @ t2 + b)
        gemm_bf16_kernel<EPI_BIAS_HSWISH, false, false><<<dim3(8, 8, 16), 256, 0, stream>>>(
            wiv, t2buf, h1buf, 1024, 256,
            inv_b, nullptr, nullptr, nullptr, nullptr, nullptr, nullptr);

        // out (+)= t2 + BN(W_pw @ hswish(dwconv3x3(h1)+b))   [dw fused into B staging]
        if (pass == 0) {
            gemm_bf16_kernel<EPI_BN_RES_OUT, true, false><<<dim3(2, 8, 16), 256, 0, stream>>>(
                wpw, h1buf, out, 256, 1024,
                pw_g, pw_b, pw_m, pw_v, t2buf, dwc_w, dwc_b);
        } else {
            gemm_bf16_kernel<EPI_BN_RES_OUT, true, true><<<dim3(2, 8, 16), 256, 0, stream>>>(
                wpw, h1buf, out, 256, 1024,
                pw_g, pw_b, pw_m, pw_v, t2buf, dwc_w, dwc_b);
        }
    }
    (void)in_sizes; (void)n_in; (void)out_size; (void)ws_size;
}
